// MoETradingTransformerWithProfit_13606456394104
// MI455X (gfx1250) — matrix operand order repaired and checked by reading, not compiled, not run on hardware
//
#include <hip/hip_runtime.h>
#include <math.h>

// ---------------- CDNA5 vector types (probe-confirmed WMMA signatures) -----
typedef __attribute__((ext_vector_type(16))) _Float16 v16h;
typedef __attribute__((ext_vector_type(8)))  _Float16 v8h;
typedef __attribute__((ext_vector_type(4)))  _Float16 v4h;
typedef __attribute__((ext_vector_type(8)))  float    v8f;

#define B_   16
#define S_   512
#define FIN_ 46
#define D_   256
#define H_   8
#define E_   8
#define DFF_ 1024
#define TOK_ (B_ * S_)   // 8192

__device__ __forceinline__ float gelu_f(float x) {
  return 0.5f * x * (1.0f + erff(x * 0.7071067811865475f));
}

__device__ __forceinline__ v16h cat8(v8h lo, v8h hi) {
  return __builtin_shufflevector(lo, hi, 0,1,2,3,4,5,6,7,8,9,10,11,12,13,14,15);
}

// ---- CDNA5 async global->LDS copy (16B per lane, tracked by ASYNCcnt) ----
// LDS byte address = low 32 bits of the flat pointer (ISA: LDS aperture maps
// addr[31:0] to the wave's LDS space).
__device__ __forceinline__ void async_g2l_16B(const _Float16* gsrc, _Float16* ldst) {
  unsigned l = (unsigned)(size_t)ldst;
  asm volatile("global_load_async_to_lds_b128 %0, %1, off"
               :: "v"(l), "v"(gsrc) : "memory");
}
__device__ __forceinline__ void wait_async_all() {
#if __has_builtin(__builtin_amdgcn_s_wait_asynccnt)
  __builtin_amdgcn_s_wait_asynccnt(0);
#else
  asm volatile("s_wait_asynccnt 0x0" ::: "memory");
#endif
}

// ===========================================================================
// WMMA GEMM on pre-converted f16 operands:
//   OUT16==0 : C(M,N)  = act(A @ W^T + bias) [* rowscale][+= accumulate]
//   OUT16==1 : C16(M,N)= act(A @ W^T + bias)            (f16 output)
// A: (M,K) f16 row-major, W: (N,K) f16 row-major. M%64==0, N%64==0, K%32==0.
// Block = 128 threads = 4 waves, each wave owns a 32x32 tile (2x2 WMMA frags).
// Tiles staged global->LDS with GLOBAL_LOAD_ASYNC_TO_LDS_B128.
// ===========================================================================
template <int ACT, int OUT16>
__global__ __launch_bounds__(128)
void gemm_wmma(const _Float16* __restrict__ A, const _Float16* __restrict__ W,
               const float* __restrict__ bias,
               float* __restrict__ C, _Float16* __restrict__ C16,
               int M, int N, int K,
               const float* __restrict__ rowscale, int rs_stride, int accumulate)
{
  __shared__ alignas(16) _Float16 As[64][32];
  __shared__ alignas(16) _Float16 Ws[64][32];

  const int t    = threadIdx.x;
  const int lane = t & 31;
  const int wv   = t >> 5;
  const int wm   = wv >> 1, wn = wv & 1;
  const int m0   = blockIdx.y * 64;
  const int n0   = blockIdx.x * 64;
  const int row  = lane & 15;          // A: M row within tile / B: N column
  const int g    = lane >> 4;          // lane half selects K sub-group

  v8f acc[2][2] = {};

  for (int k0 = 0; k0 < K; k0 += 32) {
    // Async-stage 64x32 f16 tiles of A and W into LDS (16B-chunk = 8 halfs).
    // 64 rows x 4 chunks = 256 chunks; 128 threads -> 2 chunks each.
#pragma unroll
    for (int i = 0; i < 2; ++i) {
      int c  = i * 128 + t;
      int r  = c >> 2, cc = (c & 3) * 8;
      async_g2l_16B(A + (size_t)(m0 + r) * K + k0 + cc, &As[r][cc]);
      async_g2l_16B(W + (size_t)(n0 + r) * K + k0 + cc, &Ws[r][cc]);
    }
    wait_async_all();
    __syncthreads();

    v16h af[2], bf[2];
#pragma unroll
    for (int tm = 0; tm < 2; ++tm) {
      // A 16x32 f16 operand in this code's k order (B matches): lanes0-15 row M hold K0-15; lanes16-31 hold K16-31
      const _Float16* base = &As[wm * 32 + tm * 16 + row][0];
      af[tm] = cat8(*(const v8h*)(base + g * 16),
                    *(const v8h*)(base + g * 16 + 8));
    }
#pragma unroll
    for (int tn = 0; tn < 2; ++tn) {
      // B 32x16 f16 operand in this code's k order (A matches): lanes0-15 col N hold K0-15; lanes16-31 hold K16-31
      const _Float16* base = &Ws[wn * 32 + tn * 16 + row][0];
      bf[tn] = cat8(*(const v8h*)(base + g * 16),
                    *(const v8h*)(base + g * 16 + 8));
    }
#pragma unroll
    for (int tm = 0; tm < 2; ++tm)
#pragma unroll
      for (int tn = 0; tn < 2; ++tn)
        acc[tm][tn] = __builtin_amdgcn_wmma_f32_16x16x32_f16(
            false, af[tm], false, bf[tn], (short)0, acc[tm][tn], false, false);
    __syncthreads();
  }

  // Epilogue. C layout: VGPR r -> row r + 8*(lane/16), col = lane%16.
  const int nl = lane & 15;
#pragma unroll
  for (int tm = 0; tm < 2; ++tm)
#pragma unroll
    for (int tn = 0; tn < 2; ++tn)
#pragma unroll
      for (int r = 0; r < 8; ++r) {
        int m = m0 + wm * 32 + tm * 16 + r + 8 * g;
        int n = n0 + wn * 32 + tn * 16 + nl;
        float v = acc[tm][tn][r] + (bias ? bias[n] : 0.0f);
        if (ACT == 1) v = gelu_f(v);
        if (rowscale) v *= rowscale[(size_t)m * rs_stride];
        size_t off = (size_t)m * N + n;
        if (OUT16) C16[off] = (_Float16)v;
        else if (accumulate) C[off] += v;
        else C[off] = v;
      }
}

// fp32 -> f16 conversion (n4 groups of 4 elements)
__global__ void cvt_f16(const float* __restrict__ in, _Float16* __restrict__ out, int n4)
{
  int i = blockIdx.x * 256 + threadIdx.x;
  if (i >= n4) return;
  float4 v = ((const float4*)in)[i];
  v4h o;
  o[0] = (_Float16)v.x; o[1] = (_Float16)v.y;
  o[2] = (_Float16)v.z; o[3] = (_Float16)v.w;
  ((v4h*)out)[i] = o;
}

// ===========================================================================
// Fused MHA for one (b, h, 16-query tile). Block = 64 threads = 2 waves.
// qkv (f16) rows: token * 768, cols [0:256)=Q [256:512)=K [512:768)=V.
// QK^T and PV both via WMMA; softmax normalization folded into PV epilogue.
// Output written as f16 for the out-projection GEMM.
// ===========================================================================
__global__ __launch_bounds__(64)
void attn_kernel(const _Float16* __restrict__ qkv, _Float16* __restrict__ out)
{
  __shared__ float                sc[16][512];  // raw scores (f32)
  __shared__ alignas(16) _Float16 pf[16][512];  // unnormalized exp (f16)
  __shared__ float red[16][4];
  __shared__ float rowmax[16];
  __shared__ float rowsum[16];

  const int q0 = blockIdx.x * 16;
  const int hh = blockIdx.y;
  const int b  = blockIdx.z;
  const int t    = threadIdx.x;
  const int lane = t & 31;
  const int w    = t >> 5;
  const int lr   = lane & 15;
  const int g    = lane >> 4;

  // Q fragment (A layout), held for the whole kernel. 16B-aligned f16 loads.
  const _Float16* qrow = qkv + ((size_t)(b * S_ + q0 + lr)) * 768 + hh * 32;
  v16h qa = cat8(*(const v8h*)(qrow + g * 16), *(const v8h*)(qrow + g * 16 + 8));

  // ---- scores: wave w covers key tiles [w*16, w*16+16) -> 256 keys each ----
  for (int j = 0; j < 16; ++j) {
    int tile = w * 16 + j;
    const _Float16* krow = qkv + ((size_t)(b * S_ + tile * 16 + lr)) * 768 + 256 + hh * 32;
    v16h kb = cat8(*(const v8h*)(krow + g * 16), *(const v8h*)(krow + g * 16 + 8));
    v8f a = {};
    a = __builtin_amdgcn_wmma_f32_16x16x32_f16(false, qa, false, kb, (short)0, a, false, false);
#pragma unroll
    for (int r = 0; r < 8; ++r)
      sc[r + 8 * g][tile * 16 + lr] = a[r] * 0.17677669529663687f; // 1/sqrt(32)
  }
  __syncthreads();

  // ---- softmax over 512 keys per row; store unnormalized exp as f16 ----
  {
    int rw = t >> 2, part = t & 3; // 16 rows x 4 parts of 128 cols
    float mx = -3.0e38f;
    for (int c = part * 128; c < part * 128 + 128; ++c) mx = fmaxf(mx, sc[rw][c]);
    red[rw][part] = mx;
    __syncthreads();
    if (part == 0)
      rowmax[rw] = fmaxf(fmaxf(red[rw][0], red[rw][1]), fmaxf(red[rw][2], red[rw][3]));
    __syncthreads();
    float m = rowmax[rw], s = 0.f;
    for (int c = part * 128; c < part * 128 + 128; ++c) {
      float e = __expf(sc[rw][c] - m);
      pf[rw][c] = (_Float16)e;
      s += e;
    }
    red[rw][part] = s;
    __syncthreads();
    if (part == 0)
      rowsum[rw] = red[rw][0] + red[rw][1] + red[rw][2] + red[rw][3];
    __syncthreads();
  }

  // ---- PV: wave w owns output cols [w*16, w*16+16) of dh=32 ----
  const int n0 = w * 16;
  v8f o = {};
  for (int kt = 0; kt < 16; ++kt) {
    const _Float16* base = &pf[lr][kt * 32];
    v16h pa = cat8(*(const v8h*)(base + g * 16),
                   *(const v8h*)(base + g * 16 + 8));
    v16h vb;
    int nv = n0 + lr;
#pragma unroll
    for (int i = 0; i < 16; ++i) {
      int kk = kt * 32 + g * 16 + i;
      vb[i] = qkv[((size_t)(b * S_ + kk)) * 768 + 512 + hh * 32 + nv];
    }
    o = __builtin_amdgcn_wmma_f32_16x16x32_f16(false, pa, false, vb, (short)0, o, false, false);
  }
#pragma unroll
  for (int r = 0; r < 8; ++r) {
    int m = r + 8 * g;
    size_t tok = (size_t)b * S_ + q0 + m;
    out[tok * 256 + hh * 32 + n0 + lr] = (_Float16)(o[r] / rowsum[m]);
  }
}

// ===========================================================================
// Small / elementwise kernels
// ===========================================================================
__global__ void input_proj(const float* __restrict__ x, const float* __restrict__ w,
                           const float* __restrict__ b, float* __restrict__ h)
{
  int idx = blockIdx.x * 256 + threadIdx.x;
  if (idx >= TOK_ * D_) return;
  int m = idx >> 8, d = idx & 255;
  int s = m & (S_ - 1);
  const float* xr = x + (size_t)m * FIN_;
  const float* wr = w + (size_t)d * FIN_;
  float acc = b[d];
  for (int f = 0; f < FIN_; ++f) acc += xr[f] * wr[f];
  int i2 = d >> 1;
  float div = __expf(-logf(10000.f) * (float)(2 * i2) / 256.f);
  float ang = (float)s * div;
  acc += (d & 1) ? cosf(ang) : sinf(ang);
  h[idx] = acc;
}

// wave-per-row residual + LayerNorm, in place: x = LN(x + res)
__global__ __launch_bounds__(256)
void ln_kernel(float* __restrict__ x, const float* __restrict__ res,
               const float* __restrict__ sg, const float* __restrict__ bt, int rows)
{
  int wv = threadIdx.x >> 5, lane = threadIdx.x & 31;
  int row = blockIdx.x * 8 + wv;
  if (row >= rows) return;
  float* xr = x + (size_t)row * 256;
  const float* rr = res + (size_t)row * 256;
  float v[8]; float sum = 0.f;
#pragma unroll
  for (int i = 0; i < 8; ++i) { int d = lane + i * 32; float u = xr[d] + rr[d]; v[i] = u; sum += u; }
  for (int m = 16; m >= 1; m >>= 1) sum += __shfl_xor(sum, m, 32);
  float mean = sum * (1.f / 256.f);
  float s2 = 0.f;
#pragma unroll
  for (int i = 0; i < 8; ++i) { float d = v[i] - mean; s2 += d * d; }
  for (int m = 16; m >= 1; m >>= 1) s2 += __shfl_xor(s2, m, 32);
  float rstd = rsqrtf(s2 * (1.f / 256.f) + 1e-5f);
#pragma unroll
  for (int i = 0; i < 8; ++i) { int d = lane + i * 32; xr[d] = (v[i] - mean) * rstd * sg[d] + bt[d]; }
}

__global__ void gate_kernel(const float* __restrict__ h, const float* __restrict__ gw,
                            const float* __restrict__ gb, float* __restrict__ logits)
{
  int idx = blockIdx.x * 256 + threadIdx.x;
  if (idx >= TOK_ * E_) return;
  int m = idx >> 3, e = idx & 7;
  const float* hr = h + (size_t)m * 256;
  const float* wr = gw + (size_t)e * 256;
  float acc = gb[e];
  for (int k = 0; k < 256; ++k) acc += hr[k] * wr[k];
  logits[idx] = acc;
}

__global__ void topk_kernel(const float* __restrict__ logits, float* __restrict__ wgt)
{
  int m = blockIdx.x * 256 + threadIdx.x;
  if (m >= TOK_) return;
  const float* lr = logits + (size_t)m * 8;
  int i0 = 0; float v0 = lr[0];
#pragma unroll
  for (int e = 1; e < 8; ++e) if (lr[e] > v0) { v0 = lr[e]; i0 = e; }
  int i1 = -1; float v1 = -3e38f;
#pragma unroll
  for (int e = 0; e < 8; ++e) if (e != i0 && lr[e] > v1) { v1 = lr[e]; i1 = e; }
  float p0 = 1.f / (1.f + __expf(v1 - v0));
  float p1 = 1.f - p0;
  float* wr = wgt + (size_t)m * 8;
#pragma unroll
  for (int e = 0; e < 8; ++e) wr[e] = 0.f;
  wr[i0] = p0; wr[i1] = p1;
}

__global__ __launch_bounds__(256)
void aux_partial(const float* __restrict__ logits, float* __restrict__ partial)
{
  __shared__ float sm[256][8];
  int t = threadIdx.x;
  int m = blockIdx.x * 256 + t;       // grid = 32 blocks -> exactly 8192 tokens
  const float* lr = logits + (size_t)m * 8;
  float p[8];
  float mx = lr[0];
#pragma unroll
  for (int e = 1; e < 8; ++e) mx = fmaxf(mx, lr[e]);
  float s = 0.f;
#pragma unroll
  for (int e = 0; e < 8; ++e) { p[e] = __expf(lr[e] - mx); s += p[e]; }
#pragma unroll
  for (int e = 0; e < 8; ++e) sm[t][e] = p[e] / s;
  __syncthreads();
  for (int stride = 128; stride >= 1; stride >>= 1) {
    if (t < stride)
#pragma unroll
      for (int e = 0; e < 8; ++e) sm[t][e] += sm[t + stride][e];
    __syncthreads();
  }
  if (t < 8) partial[blockIdx.x * 8 + t] = sm[0][t];
}

__global__ void aux_final(const float* __restrict__ partial, float* __restrict__ aux_out)
{
  if (threadIdx.x == 0 && blockIdx.x == 0) {
    float acc = 0.f;
    for (int e = 0; e < 8; ++e) {
      float s = 0.f;
      for (int b2 = 0; b2 < 32; ++b2) s += partial[b2 * 8 + e];
      float mean = s / (float)TOK_;
      acc += mean * mean;
    }
    aux_out[0] += 8.f * acc;
  }
}

__global__ void q_last_kernel(const float* __restrict__ h, const float* __restrict__ w,
                              const float* __restrict__ b, float* __restrict__ q)
{
  int idx = blockIdx.x * 256 + threadIdx.x;
  if (idx >= 16 * 256) return;
  int bb = idx >> 8, d = idx & 255;
  const float* hr = h + ((size_t)(bb * S_ + S_ - 1)) * 256;
  const float* wr = w + (size_t)d * 256;
  float acc = b[d];
  for (int k = 0; k < 256; ++k) acc += hr[k] * wr[k];
  q[idx] = acc;
}

__global__ __launch_bounds__(64)
void pooled_attn_kernel(const float* __restrict__ q, const float* __restrict__ kbuf,
                        const float* __restrict__ vbuf, float* __restrict__ out)
{
  __shared__ float sc[512];
  __shared__ float red[64];
  int hh = blockIdx.x, b = blockIdx.y;
  int t = threadIdx.x;
  const float* qh = q + (size_t)b * 256 + hh * 32;
  for (int s = t; s < 512; s += 64) {
    const float* kr = kbuf + ((size_t)(b * S_ + s)) * 256 + hh * 32;
    float acc = 0.f;
    for (int d = 0; d < 32; ++d) acc += qh[d] * kr[d];
    sc[s] = acc * 0.17677669529663687f;
  }
  __syncthreads();
  float mx = -3e38f;
  for (int s = t; s < 512; s += 64) mx = fmaxf(mx, sc[s]);
  red[t] = mx; __syncthreads();
  for (int st = 32; st >= 1; st >>= 1) { if (t < st) red[t] = fmaxf(red[t], red[t + st]); __syncthreads(); }
  float gmax = red[0];
  __syncthreads();
  float ls = 0.f;
  for (int s = t; s < 512; s += 64) { float e = __expf(sc[s] - gmax); sc[s] = e; ls += e; }
  red[t] = ls; __syncthreads();
  for (int st = 32; st >= 1; st >>= 1) { if (t < st) red[t] += red[t + st]; __syncthreads(); }
  float tot = red[0];
  if (t < 32) {
    float acc = 0.f;
    for (int s = 0; s < 512; ++s) acc += sc[s] * vbuf[((size_t)(b * S_ + s)) * 256 + hh * 32 + t];
    out[(size_t)b * 256 + hh * 32 + t] = acc / tot;
  }
}

__global__ void small_linear(const float* __restrict__ in, const float* __restrict__ w,
                             const float* __restrict__ b, float* __restrict__ out,
                             int M, int N, int K, int act)
{
  int idx = blockIdx.x * 64 + threadIdx.x;
  if (idx >= M * N) return;
  int m = idx / N, n = idx % N;
  const float* ir = in + (size_t)m * K;
  const float* wr = w + (size_t)n * K;
  float acc = b ? b[n] : 0.f;
  for (int k = 0; k < K; ++k) acc += ir[k] * wr[k];
  if (act) acc = gelu_f(acc);
  out[idx] = acc;
}

__global__ __launch_bounds__(32)
void small_ln(float* __restrict__ x, const float* __restrict__ sg, const float* __restrict__ bt,
              int Dd, int gelu_after)
{
  int row = blockIdx.x, lane = threadIdx.x;
  float* xr = x + (size_t)row * Dd;
  float sum = 0.f;
  for (int d = lane; d < Dd; d += 32) sum += xr[d];
  for (int m = 16; m >= 1; m >>= 1) sum += __shfl_xor(sum, m, 32);
  float mean = sum / (float)Dd;
  float s2 = 0.f;
  for (int d = lane; d < Dd; d += 32) { float u = xr[d] - mean; s2 += u * u; }
  for (int m = 16; m >= 1; m >>= 1) s2 += __shfl_xor(s2, m, 32);
  float rstd = rsqrtf(s2 / (float)Dd + 1e-5f);
  for (int d = lane; d < Dd; d += 32) {
    float v = (xr[d] - mean) * rstd * sg[d] + bt[d];
    if (gelu_after) v = gelu_f(v);
    xr[d] = v;
  }
}

// ===========================================================================
// Host orchestration
// ===========================================================================
extern "C" void kernel_launch(void* const* d_in, const int* in_sizes, int n_in,
                              void* d_out, int out_size, void* d_ws, size_t ws_size,
                              hipStream_t stream)
{
  (void)in_sizes; (void)n_in; (void)out_size; (void)ws_size;

  const float* x       = (const float*)d_in[0];
  const float* in_w    = (const float*)d_in[1];
  const float* in_b    = (const float*)d_in[2];
  const float* l_qkv_w = (const float*)d_in[3];
  const float* l_qkv_b = (const float*)d_in[4];
  const float* l_out_w = (const float*)d_in[5];
  const float* l_out_b = (const float*)d_in[6];
  const float* gate_w  = (const float*)d_in[7];
  const float* gate_b  = (const float*)d_in[8];
  const float* e_w1    = (const float*)d_in[9];
  const float* e_b1    = (const float*)d_in[10];
  const float* e_w2    = (const float*)d_in[11];
  const float* e_b2    = (const float*)d_in[12];
  const float* ln1_s   = (const float*)d_in[13];
  const float* ln1_b   = (const float*)d_in[14];
  const float* ln2_s   = (const float*)d_in[15];
  const float* ln2_b   = (const float*)d_in[16];
  const float* p_qkv_w = (const float*)d_in[17];
  const float* p_qkv_b = (const float*)d_in[18];
  const float* p_out_w = (const float*)d_in[19];
  const float* p_out_b = (const float*)d_in[20];
  const float* a_w1    = (const float*)d_in[21];
  const float* a_b1    = (const float*)d_in[22];
  const float* a_ln1s  = (const float*)d_in[23];
  const float* a_ln1b  = (const float*)d_in[24];
  const float* a_w2    = (const float*)d_in[25];
  const float* a_b2    = (const float*)d_in[26];
  const float* a_ln2s  = (const float*)d_in[27];
  const float* a_ln2b  = (const float*)d_in[28];
  const float* a_w3    = (const float*)d_in[29];
  const float* a_b3    = (const float*)d_in[30];
  const float* pr_w1   = (const float*)d_in[31];
  const float* pr_b1   = (const float*)d_in[32];
  const float* pr_ln1s = (const float*)d_in[33];
  const float* pr_ln1b = (const float*)d_in[34];
  const float* pr_w2   = (const float*)d_in[35];
  const float* pr_b2   = (const float*)d_in[36];
  const float* pr_ln2s = (const float*)d_in[37];
  const float* pr_ln2b = (const float*)d_in[38];
  const float* pr_w3   = (const float*)d_in[39];
  const float* pr_b3   = (const float*)d_in[40];

  // workspace layout
  char* ws = (char*)d_ws;
  size_t off = 0;
  auto carve = [&](size_t bytes) -> void* {
    void* p = ws + off;
    off += (bytes + 255) & ~(size_t)255;
    return p;
  };
  float*     h        = (float*)    carve((size_t)TOK_ * 256 * 4);
  _Float16*  h_h      = (_Float16*) carve((size_t)TOK_ * 256 * 2);
  _Float16*  qkv_h    = (_Float16*) carve((size_t)TOK_ * 768 * 2);
  _Float16*  attn_h   = (_Float16*) carve((size_t)TOK_ * 256 * 2);
  float*     tmp      = (float*)    carve((size_t)TOK_ * 256 * 4);
  _Float16*  hid_h    = (_Float16*) carve((size_t)TOK_ * 1024 * 2);
  float*     moe      = (float*)    carve((size_t)TOK_ * 256 * 4);
  float*     vbuf     = (float*)    carve((size_t)TOK_ * 256 * 4);
  float*     logits   = (float*)    carve((size_t)TOK_ * 8 * 4);
  float*     wgt      = (float*)    carve((size_t)TOK_ * 8 * 4);
  float*     partial  = (float*)    carve(32 * 8 * 4);
  float*     q16      = (float*)    carve(16 * 256 * 4);
  float*     pool     = (float*)    carve(16 * 256 * 4);
  float*     pool2    = (float*)    carve(16 * 256 * 4);
  float*     th1      = (float*)    carve(16 * 256 * 4);
  float*     th2      = (float*)    carve(16 * 256 * 4);
  _Float16*  qkv_w_h  = (_Float16*) carve((size_t)768 * 256 * 2);
  _Float16*  out_w_h  = (_Float16*) carve((size_t)256 * 256 * 2);
  _Float16*  p_qkv_w_h= (_Float16*) carve((size_t)768 * 256 * 2);
  _Float16*  e_w1_h   = (_Float16*) carve((size_t)E_ * DFF_ * 256 * 2);
  _Float16*  e_w2_h   = (_Float16*) carve((size_t)E_ * 256 * DFF_ * 2);

  float* out_action = (float*)d_out;       // (16,3) -> 48 floats
  float* out_profit = out_action + 48;     // (16,1) -> 16 floats
  float* out_aux    = out_action + 64;     // scalar
  hipMemsetAsync(out_aux, 0, sizeof(float), stream);

  auto cvt = [&](const float* src, _Float16* dst, size_t n) {
    cvt_f16<<<dim3((unsigned)((n / 4 + 255) / 256)), 256, 0, stream>>>(src, dst, (int)(n / 4));
  };

  // One-time weight conversions (weights shared across all 6 layers)
  cvt(l_qkv_w, qkv_w_h,   (size_t)768 * 256);
  cvt(l_out_w, out_w_h,   (size_t)256 * 256);
  cvt(p_qkv_w, p_qkv_w_h, (size_t)768 * 256);
  cvt(e_w1,    e_w1_h,    (size_t)E_ * DFF_ * 256);
  cvt(e_w2,    e_w2_h,    (size_t)E_ * 256 * DFF_);

  // input projection + positional encoding
  input_proj<<<dim3((TOK_ * 256 + 255) / 256), 256, 0, stream>>>(x, in_w, in_b, h);
  cvt(h, h_h, (size_t)TOK_ * 256);

  for (int l = 0; l < 6; ++l) {   // shared-weight encoder layers
    // QKV projection (M=8192, N=768, K=256), f16 output for attention
    gemm_wmma<0, 1><<<dim3(768 / 64, TOK_ / 64), 128, 0, stream>>>(
        h_h, qkv_w_h, l_qkv_b, nullptr, qkv_h, TOK_, 768, 256, nullptr, 0, 0);
    // fused attention (f16 in, f16 out)
    attn_kernel<<<dim3(S_ / 16, H_, B_), 64, 0, stream>>>(qkv_h, attn_h);
    // output projection (f32 out, feeds residual+LN)
    gemm_wmma<0, 0><<<dim3(256 / 64, TOK_ / 64), 128, 0, stream>>>(
        attn_h, out_w_h, l_out_b, tmp, nullptr, TOK_, 256, 256, nullptr, 0, 0);
    // h = LN(h + attn_out); refresh f16 copy of h
    ln_kernel<<<dim3(TOK_ / 8), 256, 0, stream>>>(h, tmp, ln1_s, ln1_b, TOK_);
    cvt(h, h_h, (size_t)TOK_ * 256);

    // MoE gating
    gate_kernel<<<dim3(TOK_ * 8 / 256), 256, 0, stream>>>(h, gate_w, gate_b, logits);
    topk_kernel<<<dim3(TOK_ / 256), 256, 0, stream>>>(logits, wgt);
    aux_partial<<<dim3(32), 256, 0, stream>>>(logits, partial);
    aux_final<<<dim3(1), 32, 0, stream>>>(partial, out_aux);

    // experts: moe = sum_e wgt[:,e] * (gelu(h@w1_e^T+b1_e) @ w2_e^T + b2_e)
    hipMemsetAsync(moe, 0, (size_t)TOK_ * 256 * 4, stream);
    for (int e = 0; e < 8; ++e) {
      gemm_wmma<1, 1><<<dim3(1024 / 64, TOK_ / 64), 128, 0, stream>>>(
          h_h, e_w1_h + (size_t)e * DFF_ * 256, e_b1 + (size_t)e * DFF_,
          nullptr, hid_h, TOK_, 1024, 256, nullptr, 0, 0);
      gemm_wmma<0, 0><<<dim3(256 / 64, TOK_ / 64), 128, 0, stream>>>(
          hid_h, e_w2_h + (size_t)e * 256 * DFF_, e_b2 + (size_t)e * 256,
          moe, nullptr, TOK_, 256, 1024, wgt + e, 8, 1);
    }
    // h = LN(h + moe); refresh f16 copy of h
    ln_kernel<<<dim3(TOK_ / 8), 256, 0, stream>>>(h, moe, ln2_s, ln2_b, TOK_);
    cvt(h, h_h, (size_t)TOK_ * 256);
  }

  // pooled attention (query = last token per batch)
  gemm_wmma<0, 0><<<dim3(256 / 64, TOK_ / 64), 128, 0, stream>>>(
      h_h, p_qkv_w_h + (size_t)256 * 256, p_qkv_b + 256, tmp, nullptr,
      TOK_, 256, 256, nullptr, 0, 0); // K
  gemm_wmma<0, 0><<<dim3(256 / 64, TOK_ / 64), 128, 0, stream>>>(
      h_h, p_qkv_w_h + (size_t)512 * 256, p_qkv_b + 512, vbuf, nullptr,
      TOK_, 256, 256, nullptr, 0, 0); // V
  q_last_kernel<<<dim3(16), 256, 0, stream>>>(h, p_qkv_w, p_qkv_b, q16);
  pooled_attn_kernel<<<dim3(8, 16), 64, 0, stream>>>(q16, tmp, vbuf, pool);
  small_linear<<<dim3((16 * 256 + 63) / 64), 64, 0, stream>>>(pool, p_out_w, p_out_b, pool2, 16, 256, 256, 0);

  // action head: Linear->GELU->LN->Linear->GELU->LN->Linear
  small_linear<<<dim3((16 * 128 + 63) / 64), 64, 0, stream>>>(pool2, a_w1, a_b1, th1, 16, 128, 256, 1);
  small_ln<<<dim3(16), 32, 0, stream>>>(th1, a_ln1s, a_ln1b, 128, 0);
  small_linear<<<dim3((16 * 64 + 63) / 64), 64, 0, stream>>>(th1, a_w2, a_b2, th2, 16, 64, 128, 1);
  small_ln<<<dim3(16), 32, 0, stream>>>(th2, a_ln2s, a_ln2b, 64, 0);
  small_linear<<<dim3(1), 64, 0, stream>>>(th2, a_w3, a_b3, out_action, 16, 3, 64, 0);

  // profit head: Linear->LN->GELU->Linear->LN->GELU->Linear
  small_linear<<<dim3((16 * 256 + 63) / 64), 64, 0, stream>>>(pool2, pr_w1, pr_b1, th1, 16, 256, 256, 0);
  small_ln<<<dim3(16), 32, 0, stream>>>(th1, pr_ln1s, pr_ln1b, 256, 1);
  small_linear<<<dim3((16 * 128 + 63) / 64), 64, 0, stream>>>(th1, pr_w2, pr_b2, th2, 16, 128, 256, 0);
  small_ln<<<dim3(16), 32, 0, stream>>>(th2, pr_ln2s, pr_ln2b, 128, 1);
  small_linear<<<dim3(1), 64, 0, stream>>>(th2, pr_w3, pr_b3, out_profit, 16, 1, 128, 0);
}
